// OctoTransformer_75325136437556
// MI455X (gfx1250) — matrix operand order repaired and checked by reading, not compiled, not run on hardware
//
#include <hip/hip_runtime.h>
#include <stdint.h>

// ---------------- static model config ----------------
constexpr int Bx  = 2;
constexpr int Hh  = 5;
constexpr int NT  = 16;
constexpr int NP  = 256;
constexpr int NW  = 64;
constexpr int NR  = 1;
constexpr int DIN = 512;
constexpr int D   = 768;
constexpr int NH  = 12;
constexpr int DH  = 64;
constexpr int LYR = 12;
constexpr int DF  = 3072;
constexpr int PT  = NP + NW + NR;     // 321
constexpr int S   = NT + Hh * PT;     // 1621
constexpr int M   = Bx * S;           // 3242
constexpr int MP  = 3264;             // row-padded (51 x 64)
constexpr int SP  = 1632;             // key-padded (51 x 32)

typedef __attribute__((ext_vector_type(16))) __bf16 v16bf;
typedef __attribute__((ext_vector_type(8)))  float  v8f;

#define DEV static __device__ __forceinline__

DEV uint16_t f2bf(float f) {
  uint32_t u = __float_as_uint(f);
  uint32_t r = u + 0x7FFFu + ((u >> 16) & 1u);   // round-to-nearest-even
  return (uint16_t)(r >> 16);
}

// gfx1250 async global->LDS copy (16 bytes per lane) + wait. ASYNCcnt tracked.
DEV void async_copy_b128(uint32_t lds_off, const void* gptr) {
  asm volatile("global_load_async_to_lds_b128 %0, %1, off"
               :
               : "v"(lds_off), "v"((uint64_t)(uintptr_t)gptr)
               : "memory");
}
DEV void async_wait0() { asm volatile("s_wait_asynccnt 0x0" ::: "memory"); }
DEV void compiler_fence() { asm volatile("" ::: "memory"); }

// ---------------- embedding ----------------
__global__ void k_embed(const float* __restrict__ task, const float* __restrict__ obsp,
                        const float* __restrict__ obsw,
                        const float* __restrict__ Wt, const float* __restrict__ bt,
                        const float* __restrict__ Wp, const float* __restrict__ bp,
                        const float* __restrict__ Ww, const float* __restrict__ bw,
                        const float* __restrict__ post, const float* __restrict__ posp,
                        const float* __restrict__ posw, const float* __restrict__ posr,
                        float* __restrict__ h) {
  int idx = blockIdx.x * blockDim.x + threadIdx.x;
  if (idx >= MP * D) return;
  int row = idx / D, col = idx % D;
  if (row >= M) { h[idx] = 0.f; return; }
  int b = row / S, s = row % S;
  float acc;
  if (s < NT) {
    acc = bt[col] + post[s * D + col];
    const float* src = task + (size_t)(b * NT + s) * DIN;
    for (int j = 0; j < DIN; j++) acc += src[j] * Wt[j * D + col];
  } else {
    int r = s - NT, t = r / PT, p = r % PT;
    if (p < NP) {
      acc = bp[col] + posp[(t * NP + p) * D + col];
      const float* src = obsp + (size_t)((b * Hh + t) * NP + p) * DIN;
      for (int j = 0; j < DIN; j++) acc += src[j] * Wp[j * D + col];
    } else if (p < NP + NW) {
      int pw = p - NP;
      acc = bw[col] + posw[(t * NW + pw) * D + col];
      const float* src = obsw + (size_t)((b * Hh + t) * NW + pw) * DIN;
      for (int j = 0; j < DIN; j++) acc += src[j] * Ww[j * D + col];
    } else {
      acc = posr[(t * NR + (p - NP - NW)) * D + col];
    }
  }
  h[idx] = acc;
}

// ------------- weight conversion (f32 [K,N] -> bf16 transposed [N,K]) --------
__global__ void k_cvt_bf16_t(const float* __restrict__ src, uint16_t* __restrict__ dst,
                             int K, int N) {
  int i = blockIdx.x * blockDim.x + threadIdx.x;
  if (i >= K * N) return;
  int r = i / N, c = i % N;
  dst[(size_t)c * K + r] = f2bf(src[i]);
}

__global__ void k_pack_qkv(const float* __restrict__ Wq, const float* __restrict__ Wk,
                           const float* __restrict__ Wv,
                           const float* __restrict__ bq, const float* __restrict__ bk,
                           const float* __restrict__ bv,
                           uint16_t* __restrict__ Wout, float* __restrict__ bout) {
  int i = blockIdx.x * blockDim.x + threadIdx.x;
  const int n = D * 3 * D;
  if (i < n) {
    int r = i / (3 * D), c = i % (3 * D);
    float v = (c < D) ? Wq[r * D + c] : (c < 2 * D ? Wk[r * D + c - D] : Wv[r * D + c - 2 * D]);
    Wout[(size_t)c * D + r] = f2bf(v);
  }
  if (i < 3 * D) bout[i] = (i < D) ? bq[i] : (i < 2 * D ? bk[i - D] : bv[i - 2 * D]);
}

__global__ void k_zero_u16(uint16_t* __restrict__ p, int n) {
  int i = blockIdx.x * blockDim.x + threadIdx.x;
  if (i < n) p[i] = 0;
}

// ---------------- layernorm ----------------
__global__ void __launch_bounds__(256)
k_ln(const float* __restrict__ in, const float* __restrict__ sc, const float* __restrict__ bi,
     float* __restrict__ outF, uint16_t* __restrict__ outB, int writeBF) {
  int row = blockIdx.x;
  int tid = threadIdx.x;
  const float* x = in + (size_t)row * D;
  float v[3];
  float s = 0.f;
#pragma unroll
  for (int i = 0; i < 3; i++) { v[i] = x[tid + i * 256]; s += v[i]; }
  __shared__ float red[256];
  red[tid] = s; __syncthreads();
  for (int off = 128; off > 0; off >>= 1) { if (tid < off) red[tid] += red[tid + off]; __syncthreads(); }
  float mean = red[0] * (1.0f / D);
  __syncthreads();
  s = 0.f;
#pragma unroll
  for (int i = 0; i < 3; i++) { v[i] -= mean; s += v[i] * v[i]; }
  red[tid] = s; __syncthreads();
  for (int off = 128; off > 0; off >>= 1) { if (tid < off) red[tid] += red[tid + off]; __syncthreads(); }
  float rs = rsqrtf(red[0] * (1.0f / D) + 1e-6f);
#pragma unroll
  for (int i = 0; i < 3; i++) {
    int c = tid + i * 256;
    float o = v[i] * rs * sc[c] + bi[c];
    if (writeBF) outB[(size_t)row * D + c] = f2bf(o);
    else         outF[(size_t)row * D + c] = o;
  }
}

// ---------------- bf16 WMMA GEMM: C = A[M,K] * BT[N,K]^T (+bias epilogue) -----
template <int K>
__global__ void __launch_bounds__(128)
k_gemm_bf16(const uint16_t* __restrict__ A, const uint16_t* __restrict__ BT,
            const float* __restrict__ bias, const float* __restrict__ res,
            float* __restrict__ outF, uint16_t* __restrict__ outB,
            int N, int epi) {
  constexpr int BM = 64, BN = 64, BK = 32;
  __shared__ __align__(16) uint16_t As[BM * BK];   // [m][k]
  __shared__ __align__(16) uint16_t Bs[BN * BK];   // [n][k]
  int tid  = threadIdx.x;
  int lane = tid & 31, w = tid >> 5;
  int wm = w >> 1, wn = w & 1;
  int half = lane >> 4, lr = lane & 15;
  int mBase = blockIdx.x * BM;
  int nBase = blockIdx.y * BN;

  v8f c00 = {}, c01 = {}, c10 = {}, c11 = {};

#pragma unroll 2
  for (int k0 = 0; k0 < K; k0 += BK) {
#pragma unroll
    for (int c = tid; c < 256; c += 128) {
      int rr = c >> 2, kc = (c & 3) * 8;
      async_copy_b128((uint32_t)(uintptr_t)(As + rr * BK + kc),
                      A  + (size_t)(mBase + rr) * K + k0 + kc);
      async_copy_b128((uint32_t)(uintptr_t)(Bs + rr * BK + kc),
                      BT + (size_t)(nBase + rr) * K + k0 + kc);
    }
    if (k0 + BK < K) {
      __builtin_prefetch(A  + (size_t)(mBase + (tid >> 2)) * K + k0 + BK, 0, 1);
      __builtin_prefetch(BT + (size_t)(nBase + (tid >> 2)) * K + k0 + BK, 0, 1);
    }
    async_wait0();
    __syncthreads();

    union { uint4 u[2]; v16bf v; } a0, a1, b0, b1;
    const uint16_t* ar0 = As + (wm * 32 + lr) * BK;
    const uint16_t* ar1 = As + (wm * 32 + 16 + lr) * BK;
    a0.u[0] = *(const uint4*)(ar0 + half * 8);
    a0.u[1] = *(const uint4*)(ar0 + 16 + half * 8);
    a1.u[0] = *(const uint4*)(ar1 + half * 8);
    a1.u[1] = *(const uint4*)(ar1 + 16 + half * 8);
    const uint16_t* bc0 = Bs + (wn * 32 + lr) * BK + half * 8;
    const uint16_t* bc1 = Bs + (wn * 32 + 16 + lr) * BK + half * 8;
    b0.u[0] = *(const uint4*)(bc0);
    b0.u[1] = *(const uint4*)(bc0 + 16);
    b1.u[0] = *(const uint4*)(bc1);
    b1.u[1] = *(const uint4*)(bc1 + 16);

    c00 = __builtin_amdgcn_wmma_f32_16x16x32_bf16(false, a0.v, false, b0.v, (short)0, c00, false, false);
    c01 = __builtin_amdgcn_wmma_f32_16x16x32_bf16(false, a0.v, false, b1.v, (short)0, c01, false, false);
    c10 = __builtin_amdgcn_wmma_f32_16x16x32_bf16(false, a1.v, false, b0.v, (short)0, c10, false, false);
    c11 = __builtin_amdgcn_wmma_f32_16x16x32_bf16(false, a1.v, false, b1.v, (short)0, c11, false, false);
    __syncthreads();
  }

#pragma unroll
  for (int fm = 0; fm < 2; fm++) {
#pragma unroll
    for (int fn = 0; fn < 2; fn++) {
      v8f c = fm ? (fn ? c11 : c10) : (fn ? c01 : c00);
#pragma unroll
      for (int v = 0; v < 8; v++) {
        int row = mBase + wm * 32 + fm * 16 + half * 8 + v;
        int col = nBase + wn * 32 + fn * 16 + lr;
        float val = c[v] + bias[col];
        size_t o = (size_t)row * N + col;
        if (epi == 1) {
          outF[o] = val + res[o];
        } else if (epi == 2) {
          float x3 = val * val * val;
          float g = 0.5f * val * (1.f + tanhf(0.7978845608028654f * (val + 0.044715f * x3)));
          outB[o] = f2bf(g);
        } else {
          outF[o] = val;
        }
      }
    }
  }
}

// -------- K/V prep: Kbf[bh][s][d] (bf16), Vt[bh][d][s] (bf16, transposed) ----
__global__ void k_reshape_kv(const float* __restrict__ qkv,
                             uint16_t* __restrict__ Kbf, uint16_t* __restrict__ Vt) {
  int i = blockIdx.x * blockDim.x + threadIdx.x;
  const int total = Bx * NH * S * DH;
  if (i >= total) return;
  int d = i % DH; int t = i / DH;
  int s = t % S;  t /= S;
  int h = t % NH; int b = t / NH;
  const float* base = qkv + (size_t)(b * S + s) * (3 * D);
  int bh = b * NH + h;
  Kbf[((size_t)bh * SP + s) * DH + d] = f2bf(base[D + h * DH + d]);
  Vt [((size_t)bh * DH + d) * SP + s] = f2bf(base[2 * D + h * DH + d]);
}

// ---------------- attention mask helpers ----------------
DEV void decode_tok(int s, int& g, int& t) {
  if (s < NT) { g = 0; t = -1; }
  else {
    int r = s - NT; t = r / PT; int p = r % PT;
    g = (p < NP) ? 1 : ((p < NP + NW) ? 2 : 3);
  }
}
DEV bool attn_allow(int gq, int tq, int sk, const unsigned char* pad, int b) {
  int gk, tk; decode_tok(sk, gk, tk);
  bool r;
  if (gq == 0)      r = (gk == 0);
  else if (gq == 3) r = true;
  else              r = (gk != 3);
  if (!r) return false;
  if (tk > tq) return false;
  if ((gk == 1 || gk == 2) && pad[b * Hh + tk] == 0) return false;
  return true;
}

// ------------- WMMA flash attention: 4 waves/block, 16 queries/wave ----------
__global__ void __launch_bounds__(128)
k_attn_wmma(const float* __restrict__ qkv, const uint16_t* __restrict__ Kbf,
            const uint16_t* __restrict__ Vt, const unsigned char* __restrict__ pad,
            uint16_t* __restrict__ attb) {
  __shared__ __align__(16) uint16_t Ks[32 * 64];      // [key][dim]  (shared by 4 waves)
  __shared__ __align__(16) uint16_t Vs[64 * 32];      // [dim][key]
  __shared__ __align__(16) uint16_t Qs[4][16 * 64];   // per-wave query tile
  __shared__ __align__(16) uint16_t Ps[4][16 * 32];   // per-wave P tile (C->A relayout)

  int tid = threadIdx.x, lane = tid & 31, w = tid >> 5;
  int half = lane >> 4, lr = lane & 15;
  int hh = blockIdx.y, b = blockIdx.z;
  int bh = b * NH + hh;
  int qt = blockIdx.x * 4 + w;
  if (qt > SP / 16 - 1) qt = SP / 16 - 1;

  // stage Q tile (pre-scaled, row-clamped) into LDS as bf16
  for (int i = lane; i < 16 * 64; i += 32) {
    int r = i >> 6, d = i & 63;
    int sq = qt * 16 + r; if (sq >= S) sq = S - 1;
    Qs[w][i] = f2bf(qkv[(size_t)(b * S + sq) * (3 * D) + hh * DH + d] * 0.125f);
  }
  // per-lane row metadata (8 rows each: half*8+v)
  int gqv[8], tqv[8];
#pragma unroll
  for (int v = 0; v < 8; v++) {
    int sq = qt * 16 + half * 8 + v; if (sq >= S) sq = S - 1;
    decode_tok(sq, gqv[v], tqv[v]);
  }
  compiler_fence();
  union { uint4 u[2]; v16bf v; } aq0, aq1;
  {
    const uint16_t* qr = &Qs[w][lr * 64];
    aq0.u[0] = *(const uint4*)(qr + half * 8);
    aq0.u[1] = *(const uint4*)(qr + 16 + half * 8);
    aq1.u[0] = *(const uint4*)(qr + 32 + half * 8);
    aq1.u[1] = *(const uint4*)(qr + 32 + 16 + half * 8);
  }

  v8f o0 = {}, o1 = {}, o2 = {}, o3 = {};
  float mrun[8], lrun[8];
#pragma unroll
  for (int v = 0; v < 8; v++) { mrun[v] = -1e9f; lrun[v] = 0.f; }

  const uint16_t* Kb = Kbf + (size_t)bh * SP * DH;
  const uint16_t* Vb = Vt  + (size_t)bh * DH * SP;

  for (int kb = 0; kb < SP; kb += 32) {
    // cooperative async tile loads (512 x 16B chunks, 128 threads)
#pragma unroll
    for (int c = tid; c < 256; c += 128) {
      int key = c >> 3, dc = (c & 7) * 8;
      async_copy_b128((uint32_t)(uintptr_t)(Ks + key * 64 + dc),
                      Kb + (size_t)(kb + key) * DH + dc);
      int dim = c >> 2, kc = (c & 3) * 8;
      async_copy_b128((uint32_t)(uintptr_t)(Vs + dim * 32 + kc),
                      Vb + (size_t)dim * SP + kb + kc);
    }
    async_wait0();
    __syncthreads();

    // ---- S-tile = Q x K^T : 2 key halves x 2 dim slices = 4 WMMA ----
    v8f s0 = {}, s1 = {};
#pragma unroll
    for (int dsl = 0; dsl < 2; dsl++) {
      union { uint4 u[2]; v16bf v; } bk0, bk1;
      const uint16_t* k0p = Ks + lr * 64 + dsl * 32 + half * 8;
      const uint16_t* k1p = Ks + (16 + lr) * 64 + dsl * 32 + half * 8;
      bk0.u[0] = *(const uint4*)(k0p); bk0.u[1] = *(const uint4*)(k0p + 16);
      bk1.u[0] = *(const uint4*)(k1p); bk1.u[1] = *(const uint4*)(k1p + 16);
      const v16bf aq = dsl ? aq1.v : aq0.v;
      s0 = __builtin_amdgcn_wmma_f32_16x16x32_bf16(false, aq, false, bk0.v, (short)0, s0, false, false);
      s1 = __builtin_amdgcn_wmma_f32_16x16x32_bf16(false, aq, false, bk1.v, (short)0, s1, false, false);
    }

    // ---- mask + online softmax on C-fragment layout ----
#pragma unroll
    for (int v = 0; v < 8; v++) {
      int key0 = kb + lr, key1 = kb + 16 + lr;
      float x0 = (key0 < S && attn_allow(gqv[v], tqv[v], key0, pad, b)) ? s0[v] : -1e9f;
      float x1 = (key1 < S && attn_allow(gqv[v], tqv[v], key1, pad, b)) ? s1[v] : -1e9f;
      float tm = fmaxf(x0, x1);
#pragma unroll
      for (int off = 8; off > 0; off >>= 1) tm = fmaxf(tm, __shfl_xor(tm, off, 32));
      float mnew = fmaxf(mrun[v], tm);
      float scale = __expf(mrun[v] - mnew);
      float p0 = __expf(x0 - mnew), p1 = __expf(x1 - mnew);
      float rs = p0 + p1;
#pragma unroll
      for (int off = 8; off > 0; off >>= 1) rs += __shfl_xor(rs, off, 32);
      lrun[v] = lrun[v] * scale + rs;
      mrun[v] = mnew;
      o0[v] *= scale; o1[v] *= scale; o2[v] *= scale; o3[v] *= scale;
      Ps[w][(half * 8 + v) * 32 + lr]      = f2bf(p0);
      Ps[w][(half * 8 + v) * 32 + 16 + lr] = f2bf(p1);
    }
    compiler_fence();   // DS ops are wave-in-order; just stop compiler reordering

    // ---- O += P x V : 4 WMMA (N=64, K=32) ----
    union { uint4 u[2]; v16bf v; } ap;
    const uint16_t* pr = &Ps[w][lr * 32];
    ap.u[0] = *(const uint4*)(pr + half * 8);
    ap.u[1] = *(const uint4*)(pr + 16 + half * 8);
#pragma unroll
    for (int f = 0; f < 4; f++) {
      union { uint4 u[2]; v16bf v; } bv_;
      const uint16_t* vp = Vs + (f * 16 + lr) * 32 + half * 8;
      bv_.u[0] = *(const uint4*)(vp); bv_.u[1] = *(const uint4*)(vp + 16);
      v8f& o = f == 0 ? o0 : (f == 1 ? o1 : (f == 2 ? o2 : o3));
      o = __builtin_amdgcn_wmma_f32_16x16x32_bf16(false, ap.v, false, bv_.v, (short)0, o, false, false);
    }
    __syncthreads();
  }

  // ---- normalize + store ----
#pragma unroll
  for (int v = 0; v < 8; v++) {
    float inv = 1.f / lrun[v];
    int sq = qt * 16 + half * 8 + v;
    if (sq < S) {
      uint16_t* orow = attb + (size_t)(b * S + sq) * D + hh * DH;
#pragma unroll
      for (int f = 0; f < 4; f++) {
        v8f& o = f == 0 ? o0 : (f == 1 ? o1 : (f == 2 ? o2 : o3));
        orow[f * 16 + lr] = f2bf(o[v] * inv);
      }
    }
  }
}

// ---------------- host orchestration ----------------
extern "C" void kernel_launch(void* const* d_in, const int* in_sizes, int n_in,
                              void* d_out, int out_size, void* d_ws, size_t ws_size,
                              hipStream_t stream) {
  (void)in_sizes; (void)n_in; (void)out_size; (void)ws_size;

  const float* task = (const float*)d_in[0];
  const float* obsp = (const float*)d_in[1];
  const float* obsw = (const float*)d_in[2];
  const float* Wt   = (const float*)d_in[3];
  const float* bt   = (const float*)d_in[4];
  const float* Wp   = (const float*)d_in[5];
  const float* bp   = (const float*)d_in[6];
  const float* Ww   = (const float*)d_in[7];
  const float* bw   = (const float*)d_in[8];
  const float* post = (const float*)d_in[9];
  const float* posp = (const float*)d_in[10];
  const float* posw = (const float*)d_in[11];
  const float* posr = (const float*)d_in[12];
  const float* ln1s = (const float*)d_in[13];
  const float* ln1b = (const float*)d_in[14];
  const float* Wq   = (const float*)d_in[15];
  const float* bq   = (const float*)d_in[16];
  const float* Wk   = (const float*)d_in[17];
  const float* bk   = (const float*)d_in[18];
  const float* Wv   = (const float*)d_in[19];
  const float* bv   = (const float*)d_in[20];
  const float* Wo   = (const float*)d_in[21];
  const float* bo   = (const float*)d_in[22];
  const float* ln2s = (const float*)d_in[23];
  const float* ln2b = (const float*)d_in[24];
  const float* W1   = (const float*)d_in[25];
  const float* b1   = (const float*)d_in[26];
  const float* W2   = (const float*)d_in[27];
  const float* b2   = (const float*)d_in[28];
  const float* lnfs = (const float*)d_in[29];
  const float* lnfb = (const float*)d_in[30];
  const unsigned char* pad = (const unsigned char*)d_in[31];

  char* ws = (char*)d_ws;
  size_t off = 0;
  auto take = [&](size_t bytes) -> char* {
    char* p = ws + off;
    off = (off + bytes + 255) & ~(size_t)255;
    return p;
  };
  float*    h    = (float*)   take((size_t)MP * D * 4);
  uint16_t* ybf  = (uint16_t*)take((size_t)MP * D * 2);
  float*    qkv  = (float*)   take((size_t)MP * 3 * D * 4);
  uint16_t* attb = (uint16_t*)take((size_t)MP * D * 2);
  uint16_t* fc1  = (uint16_t*)take((size_t)MP * DF * 2);
  uint16_t* Kbf  = (uint16_t*)take((size_t)Bx * NH * SP * DH * 2);
  uint16_t* Vt   = (uint16_t*)take((size_t)Bx * NH * DH * SP * 2);
  uint16_t* wqkv = (uint16_t*)take((size_t)D * 3 * D * 2);
  uint16_t* wo   = (uint16_t*)take((size_t)D * D * 2);
  uint16_t* w1   = (uint16_t*)take((size_t)D * DF * 2);
  uint16_t* w2   = (uint16_t*)take((size_t)DF * D * 2);
  float*    bqkv = (float*)   take((size_t)3 * D * 4);

  {
    int n = MP * D;
    k_embed<<<dim3((n + 255) / 256), dim3(256), 0, stream>>>(
        task, obsp, obsw, Wt, bt, Wp, bp, Ww, bw, post, posp, posw, posr, h);
  }
  {
    int n = (MP - M) * D;
    k_zero_u16<<<dim3((n + 255) / 256), dim3(256), 0, stream>>>(attb + (size_t)M * D, n);
  }
  {
    int n = Bx * NH * SP * DH;   // zero K/V pads once; real parts rewritten per layer
    k_zero_u16<<<dim3((n + 255) / 256), dim3(256), 0, stream>>>(Kbf, n);
    k_zero_u16<<<dim3((n + 255) / 256), dim3(256), 0, stream>>>(Vt, n);
  }

  for (int l = 0; l < LYR; l++) {
    const float* Wql = Wq + (size_t)l * D * D;
    const float* Wkl = Wk + (size_t)l * D * D;
    const float* Wvl = Wv + (size_t)l * D * D;
    const float* Wol = Wo + (size_t)l * D * D;
    const float* W1l = W1 + (size_t)l * D * DF;
    const float* W2l = W2 + (size_t)l * DF * D;

    k_pack_qkv<<<dim3((D * 3 * D + 255) / 256), dim3(256), 0, stream>>>(
        Wql, Wkl, Wvl, bq + l * D, bk + l * D, bv + l * D, wqkv, bqkv);
    k_cvt_bf16_t<<<dim3((D * D + 255) / 256), dim3(256), 0, stream>>>(Wol, wo, D, D);
    k_cvt_bf16_t<<<dim3((D * DF + 255) / 256), dim3(256), 0, stream>>>(W1l, w1, D, DF);
    k_cvt_bf16_t<<<dim3((DF * D + 255) / 256), dim3(256), 0, stream>>>(W2l, w2, DF, D);

    k_ln<<<dim3(MP), dim3(256), 0, stream>>>(h, ln1s + l * D, ln1b + l * D, nullptr, ybf, 1);
    k_gemm_bf16<D><<<dim3(MP / 64, (3 * D) / 64), dim3(128), 0, stream>>>(
        ybf, wqkv, bqkv, nullptr, qkv, nullptr, 3 * D, 0);
    {
      int n = Bx * NH * S * DH;
      k_reshape_kv<<<dim3((n + 255) / 256), dim3(256), 0, stream>>>(qkv, Kbf, Vt);
    }
    k_attn_wmma<<<dim3(SP / 16 / 4 + 1, NH, Bx), dim3(128), 0, stream>>>(
        qkv, Kbf, Vt, pad, attb);
    k_gemm_bf16<D><<<dim3(MP / 64, D / 64), dim3(128), 0, stream>>>(
        attb, wo, bo + l * D, h, h, nullptr, D, 1);
    k_ln<<<dim3(MP), dim3(256), 0, stream>>>(h, ln2s + l * D, ln2b + l * D, nullptr, ybf, 1);
    k_gemm_bf16<D><<<dim3(MP / 64, DF / 64), dim3(128), 0, stream>>>(
        ybf, w1, b1 + (size_t)l * DF, nullptr, nullptr, fc1, DF, 2);
    k_gemm_bf16<DF><<<dim3(MP / 64, D / 64), dim3(128), 0, stream>>>(
        fc1, w2, b2 + l * D, h, h, nullptr, D, 1);
  }

  k_ln<<<dim3(M), dim3(256), 0, stream>>>(h, lnfs, lnfb, (float*)d_out, nullptr, 0);
}
